// SSTA_13632226197745
// MI455X (gfx1250) — compile-verified, matrix operand order checked by reading, not run on hardware
//
#include <hip/hip_runtime.h>
#include <hip/hip_bf16.h>

// ---------------------------------------------------------------------------
// RetNet-ish block for MI455X (gfx1250, wave32, WMMA f32_16x16x32_f16)
//   x:[512,64,256] -> QKV proj -> rotary -> per-(h,n) attention over S=512
//   -> LN1 -> FFN(256->1024->256, relu) -> LN2 -> out [512,64,256] f32
// GEMMs: LDS-staged B panel (dedup 8x across waves), double-buffered with
// async-to-LDS DMA when available (ASYNCcnt path), 32x64 per-wave tiles.
// ---------------------------------------------------------------------------

typedef __attribute__((ext_vector_type(8)))  _Float16 half8;
typedef __attribute__((ext_vector_type(16))) _Float16 v16h;
typedef __attribute__((ext_vector_type(8)))  float    v8f;

// exact param type per hipcc diagnostic: "__vector_size__(4*sizeof(int)) int __device__ *"
typedef int v4i_vs __attribute__((__vector_size__(16)));
typedef __attribute__((address_space(1))) v4i_vs* as1_v4i_p;
typedef __attribute__((address_space(3))) v4i_vs* as3_v4i_p;

#define S_LEN 512
#define NTOK  64
#define CDIM  256
#define HEADS 8
#define DHEAD 32
#define FDIM  1024
#define MROWS (S_LEN * NTOK)   // 32768 rows for the big GEMMs

// async global->LDS copy (gfx1250 GLOBAL_LOAD_ASYNC_TO_LDS_B128, ASYNCcnt)
#if __has_builtin(__builtin_amdgcn_global_load_async_to_lds_b128) && \
    __has_builtin(__builtin_amdgcn_s_wait_asynccnt)
#define ASYNC_CP 1
#else
#define ASYNC_CP 0
#endif

union FragU { v16h v; half8 h[2]; };

// CDNA5 16-bit A/B fragment layout (ISA 7.12.2): lane half hh=0 holds
// K = {k..k+7, k+16..k+23}; hh=1 holds K = {k+8..k+15, k+24..k+31}.
__device__ __forceinline__ v16h load_frag(const _Float16* p, int k, int hh) {
    FragU u;
    u.h[0] = *(const half8*)(p + k + 8 * hh);
    u.h[1] = *(const half8*)(p + k + 16 + 8 * hh);
    return u.v;
}

__device__ __forceinline__ float wave_sum(float v) {
#pragma unroll
    for (int o = 16; o > 0; o >>= 1) v += __shfl_xor(v, o, 32);
    return v;
}
__device__ __forceinline__ float wave_max(float v) {
#pragma unroll
    for (int o = 16; o > 0; o >>= 1) v = fmaxf(v, __shfl_xor(v, o, 32));
    return v;
}

// ---------------------------------------------------------------------------
__global__ void cvt_f32_f16(const float* __restrict__ in, _Float16* __restrict__ out, int n) {
    int i = blockIdx.x * 256 + threadIdx.x;
    if (i < n) out[i] = (_Float16)in[i];
}

// ---------------------------------------------------------------------------
// GEMM: out[m][co] = act( A[m][:] . Bw[co][:] + bias[co] )
// A: MxK f16 row-major (global); Bw: NoutxK f16 row-major.
// Block: 256 thr = 8 waves; wave: 32 rows x 64 cols (8 acc); block: 256x64.
// B panel staged in LDS, double-buffered in 64-K chunks.
#define KCHUNK 64
#define BSTRIDE (KCHUNK + 8)   // 72 halfs = 144B row stride (bank spread)

__global__ __launch_bounds__(256)
void gemm_wmma(const _Float16* __restrict__ A, const _Float16* __restrict__ Bw,
               const float* __restrict__ bias, _Float16* __restrict__ outH,
               float* __restrict__ outF, int K, int Nout, int relu)
{
    __shared__ _Float16 bsm[2][64 * BSTRIDE];       // 2 x 9KB
    const int tid  = threadIdx.x;
    const int wave = tid >> 5, lane = tid & 31;
    const int hh = lane >> 4, l16 = lane & 15;
    const int row_base = blockIdx.x * 256 + wave * 32;
    const int col_base = blockIdx.y * 64;

    const _Float16* Arow0 = A + (size_t)(row_base + l16) * K;
    const _Float16* Arow1 = Arow0 + (size_t)16 * K;

    const v8f vzero = {};
    v8f acc[2][4];
#pragma unroll
    for (int i = 0; i < 2; ++i)
#pragma unroll
        for (int t = 0; t < 4; ++t) acc[i][t] = vzero;

    // cooperative stage of one 64x64-halfs B chunk: 512 x 16B units, 2/thread
    auto stage = [&](int chunk, int buf) {
#pragma unroll
        for (int u = 0; u < 2; ++u) {
            int unit = tid + u * 256;
            int r  = unit >> 3;            // 0..63  (B row = output column)
            int c8 = (unit & 7) * 8;       // 0..56  halfs within chunk
            const _Float16* g = Bw + (size_t)(col_base + r) * K + chunk * KCHUNK + c8;
            _Float16* l = &bsm[buf][r * BSTRIDE + c8];
#if ASYNC_CP
            __builtin_amdgcn_global_load_async_to_lds_b128(
                (as1_v4i_p)g, (as3_v4i_p)l, 0, 0);
#else
            *(half8*)l = *(const half8*)g;
#endif
        }
    };

    const int nchunks = K / KCHUNK;
    stage(0, 0);
#if ASYNC_CP
    __builtin_amdgcn_s_wait_asynccnt(0);
#endif
    __syncthreads();

    for (int c = 0; c < nchunks; ++c) {
        const int buf = c & 1;
        if (c + 1 < nchunks) stage(c + 1, buf ^ 1);   // DMA overlaps compute

#pragma unroll
        for (int ks = 0; ks < KCHUNK; ks += 32) {
            const int k = c * KCHUNK + ks;
            if (k + 64 < K) __builtin_prefetch(Arow0 + k + 64, 0, 1);
            v16h a0 = load_frag(Arow0, k, hh);
            v16h a1 = load_frag(Arow1, k, hh);
#pragma unroll
            for (int t = 0; t < 4; ++t) {
                const _Float16* bb = &bsm[buf][(t * 16 + l16) * BSTRIDE];
                v16h b = load_frag(bb, ks, hh);       // ds_load_b128 x2
                acc[0][t] = __builtin_amdgcn_wmma_f32_16x16x32_f16(
                    false, a0, false, b, (short)0, acc[0][t], false, false);
                acc[1][t] = __builtin_amdgcn_wmma_f32_16x16x32_f16(
                    false, a1, false, b, (short)0, acc[1][t], false, false);
            }
        }
#if ASYNC_CP
        __builtin_amdgcn_s_wait_asynccnt(0);
        __syncthreads();
#else
        __syncthreads();                               // done reading bsm[buf]
        // fallback already wrote bsm[buf^1] via VGPRs before the barrier above;
        // a second barrier makes the new buffer visible to all waves.
        __syncthreads();
#endif
    }

#pragma unroll
    for (int rt = 0; rt < 2; ++rt) {
#pragma unroll
        for (int t = 0; t < 4; ++t) {
            int cc = col_base + t * 16 + l16;
            float bv = bias ? bias[cc] : 0.f;
#pragma unroll
            for (int r = 0; r < 8; ++r) {
                int mm = row_base + rt * 16 + r + 8 * hh;   // D: M = r + 8*(lane/16)
                float v = acc[rt][t][r] + bv;
                if (relu) v = fmaxf(v, 0.f);
                size_t idx = (size_t)mm * Nout + cc;
                if (outH) outH[idx] = (_Float16)v;
                if (outF) outF[idx] = v;
            }
        }
    }
}

// ---------------------------------------------------------------------------
// Rotary + scatter: y{q,k,v}[m= s*64+n][h*32+dd]  ->
//   qbuf/kbuf [h][n][s][dd] (rotated),  vT [h][n][dd][s] (transposed for B frags)
__global__ void rotary_scatter(const _Float16* __restrict__ yq, const _Float16* __restrict__ yk,
                               const _Float16* __restrict__ yv, _Float16* __restrict__ qb,
                               _Float16* __restrict__ kb, _Float16* __restrict__ vT)
{
    int idx = blockIdx.x * 256 + threadIdx.x;
    int dd = idx & 31;
    int s  = (idx >> 5) & 511;
    int n  = (idx >> 14) & 63;
    int h  = idx >> 20;
    if (h >= HEADS) return;

    size_t base = ((size_t)s * NTOK + n) * CDIM + h * DHEAD;  // row of y*, head slice
    // angle = repeat(1/10000^linspace(0,1,16), 2)
    float ang = powf(10000.f, -(float)(dd >> 1) / 15.f);
    float sn = sinf((float)s * ang), cs = cosf((float)s * ang);
    int p; float sgn;
    if (dd < 16) { p = 2 * dd + 1;  sgn = -1.f; }   // x2 = -x[1::2]
    else         { p = 2 * dd - 32; sgn =  1.f; }   // x2 =  x[::2]

    size_t qo = ((size_t)(h * NTOK + n) * S_LEN + s) * DHEAD + dd;
    {
        float xv = (float)yq[base + dd];
        float x2 = sgn * (float)yq[base + p];
        qb[qo] = (_Float16)(xv * cs + x2 * sn);
    }
    {
        float xv = (float)yk[base + dd];
        float x2 = sgn * (float)yk[base + p];
        kb[qo] = (_Float16)(xv * cs + x2 * sn);
    }
    vT[((size_t)(h * NTOK + n) * DHEAD + dd) * S_LEN + s] = yv[base + dd];
}

// ---------------------------------------------------------------------------
// Attention: block = (qchunk 0..7, n 0..63, h 0..7); 64 query rows, full S=512.
// 128KB dynamic LDS score tile; d=32 => ONE wmma per 16x16 score tile.
__global__ __launch_bounds__(256)
void attn_wmma(const _Float16* __restrict__ qb, const _Float16* __restrict__ kb,
               const _Float16* __restrict__ vT, float* __restrict__ attn_out)
{
    extern __shared__ float sc[];                    // [64][512] f32
    const int wave = threadIdx.x >> 5, lane = threadIdx.x & 31;
    const int hh = lane >> 4, l16 = lane & 15;
    const int qc = blockIdx.x, n = blockIdx.y, h = blockIdx.z;
    const size_t hb = (size_t)(h * NTOK + n);
    const _Float16* qbase = qb + hb * S_LEN * DHEAD;
    const _Float16* kbase = kb + hb * S_LEN * DHEAD;
    const _Float16* vbase = vT + hb * DHEAD * S_LEN;
    const float decay = logf(1.f - exp2f(-1.f - 3.f * (float)h / 8.f));

    // ---- phase 1: scores = q k^T + |i-j|*decay -> LDS
    {
        const int mt = wave & 3, nh = wave >> 2;          // 4 M-tiles x 2 col-halves
        const int sq_lane = qc * 64 + mt * 16 + l16;
        v16h a = load_frag(qbase + (size_t)sq_lane * DHEAD, 0, hh);
        v16h b = load_frag(kbase + (size_t)(nh * 256 + l16) * DHEAD, 0, hh);
#pragma unroll 4
        for (int nt = 0; nt < 16; ++nt) {
            int sk = nh * 256 + nt * 16 + l16;
            v16h bn;                                      // prefetch next B frag
            if (nt + 1 < 16)
                bn = load_frag(kbase + (size_t)(sk + 16) * DHEAD, 0, hh);
            v8f d = {};
            d = __builtin_amdgcn_wmma_f32_16x16x32_f16(
                false, a, false, b, (short)0, d, false, false);
#pragma unroll
            for (int r = 0; r < 8; ++r) {
                int rloc = mt * 16 + r + 8 * hh;
                int sqr  = qc * 64 + rloc;
                sc[rloc * S_LEN + sk] = d[r] + fabsf((float)(sqr - sk)) * decay;
            }
            b = bn;
        }
    }
    __syncthreads();

    // ---- phase 2: row softmax (wave32 shuffle reductions, 8 rows/wave)
    for (int rr = 0; rr < 8; ++rr) {
        float* row = sc + (wave * 8 + rr) * S_LEN;
        float mx = -1e30f;
        for (int j = lane; j < S_LEN; j += 32) mx = fmaxf(mx, row[j]);
        mx = wave_max(mx);
        float sm = 0.f;
        for (int j = lane; j < S_LEN; j += 32) { float e = expf(row[j] - mx); row[j] = e; sm += e; }
        sm = wave_sum(sm);
        float inv = 1.f / sm;
        for (int j = lane; j < S_LEN; j += 32) row[j] *= inv;
    }
    __syncthreads();

    // ---- phase 3: O = P V  (P from LDS f32 -> f16 frags; V pre-transposed)
    {
        const int mt = wave & 3, nt2 = wave >> 2;         // 4 M-tiles x 2 dd-tiles
        const int mloc = mt * 16 + l16;
        const int dd = nt2 * 16 + l16;
        const _Float16* vrow = vbase + (size_t)dd * S_LEN;
        v8f acc = {};
        for (int kk = 0; kk < S_LEN; kk += 32) {
            FragU u;
            const float* prow = sc + mloc * S_LEN + kk + 8 * hh;
#pragma unroll
            for (int i = 0; i < 8; ++i) {
                u.h[0][i] = (_Float16)prow[i];
                u.h[1][i] = (_Float16)prow[16 + i];
            }
            v16h b = load_frag(vrow, kk, hh);
            acc = __builtin_amdgcn_wmma_f32_16x16x32_f16(
                false, u.v, false, b, (short)0, acc, false, false);
        }
#pragma unroll
        for (int r = 0; r < 8; ++r) {
            int sqr = qc * 64 + mt * 16 + r + 8 * hh;
            attn_out[((size_t)n * S_LEN + sqr) * CDIM + h * DHEAD + nt2 * 16 + l16] = acc[r];
        }
    }
}

// ---------------------------------------------------------------------------
// Residual + LayerNorm. Rows m = n*512 + s. mode 0: residual read from x
// ([s][n][c] layout), outputs f32+f16 in m-layout (for FFN). mode 1: residual
// in m-layout, f32 output written transposed back to [s][n][c] (= d_out).
__global__ __launch_bounds__(256)
void ln_kernel(const float* __restrict__ A, const float* __restrict__ R,
               const float* __restrict__ g, const float* __restrict__ be,
               float* __restrict__ out32, _Float16* __restrict__ out16, int mode)
{
    int wave = threadIdx.x >> 5, lane = threadIdx.x & 31;
    size_t m = (size_t)blockIdx.x * 8 + wave;
    int n = (int)(m >> 9), s = (int)(m & 511);
    size_t mrow = m * CDIM;
    size_t trow = ((size_t)s * NTOK + n) * CDIM;
    const float* arow = A + mrow;

    float vals[8]; float sum = 0.f;
#pragma unroll
    for (int i = 0; i < 8; ++i) {
        int c = lane + 32 * i;
        float res = (mode == 0) ? R[trow + c] : R[mrow + c];
        float v = arow[c] + res;
        vals[i] = v; sum += v;
    }
    sum = wave_sum(sum);
    float mu = sum * (1.f / 256.f);
    float vv = 0.f;
#pragma unroll
    for (int i = 0; i < 8; ++i) { float d = vals[i] - mu; vv += d * d; }
    vv = wave_sum(vv);
    float inv = rsqrtf(vv * (1.f / 256.f) + 1e-5f);
#pragma unroll
    for (int i = 0; i < 8; ++i) {
        int c = lane + 32 * i;
        float o = (vals[i] - mu) * inv * g[c] + be[c];
        if (mode == 0) { out32[mrow + c] = o; out16[mrow + c] = (_Float16)o; }
        else           { out32[trow + c] = o; }
    }
}

// ---------------------------------------------------------------------------
extern "C" void kernel_launch(void* const* d_in, const int* in_sizes, int n_in,
                              void* d_out, int out_size, void* d_ws, size_t ws_size,
                              hipStream_t stream) {
    (void)in_sizes; (void)n_in; (void)out_size; (void)ws_size;
    const float* x    = (const float*)d_in[0];
    const float* wq   = (const float*)d_in[1];
    const float* bq   = (const float*)d_in[2];
    const float* wk   = (const float*)d_in[3];
    const float* bk   = (const float*)d_in[4];
    const float* wv   = (const float*)d_in[5];
    const float* bv   = (const float*)d_in[6];
    const float* ln1w = (const float*)d_in[7];
    const float* ln1b = (const float*)d_in[8];
    const float* ln2w = (const float*)d_in[9];
    const float* ln2b = (const float*)d_in[10];
    const float* w1   = (const float*)d_in[11];
    const float* b1   = (const float*)d_in[12];
    const float* w2   = (const float*)d_in[13];
    const float* b2   = (const float*)d_in[14];

    const size_t MB = 1ull << 20;
    char* ws = (char*)d_ws;
    // Aliased scratch plan (peak ~162 MB), lifetimes verified in sequence below.
    _Float16* wq16 = (_Float16*)(ws);             // 128KB
    _Float16* wk16 = wq16 + 65536;
    _Float16* wv16 = wk16 + 65536;
    _Float16* w116 = wv16 + 65536;                // 512KB
    _Float16* w216 = w116 + 262144;               // 512KB
    _Float16* x16  = (_Float16*)(ws + 2  * MB);   // 16MB  [ 2..18)
    _Float16* yq   = (_Float16*)(ws + 18 * MB);   // 16MB  [18..34)
    _Float16* yk   = (_Float16*)(ws + 34 * MB);   // 16MB  [34..50)
    _Float16* yv   = (_Float16*)(ws + 50 * MB);   // 16MB  [50..66)
    _Float16* qbuf = (_Float16*)(ws + 66 * MB);   // 16MB  [66..82)
    _Float16* kbuf = (_Float16*)(ws + 82 * MB);   // 16MB  [82..98)
    _Float16* vTb  = (_Float16*)(ws + 98 * MB);   // 16MB  [98..114)
    float*    attn = (float*)   (ws + 18 * MB);   // 32MB  [18..50)  reuse: y* dead after rotary
    _Float16* ln1h = (_Float16*)(ws + 50 * MB);   // 16MB  [50..66)  reuse: yv dead
    float*    ln1f = (float*)   (ws + 66 * MB);   // 32MB  [66..98)  reuse: q/k dead after attn
    _Float16* h1   = (_Float16*)(ws + 98 * MB);   // 64MB  [98..162) reuse: vT dead
    float*    ffn  = (float*)   (ws + 18 * MB);   // 32MB  [18..50)  reuse: attn dead after LN1

    // 1) f32 -> f16 conversions
    cvt_f32_f16<<<(MROWS * CDIM + 255) / 256, 256, 0, stream>>>(x, x16, MROWS * CDIM);
    cvt_f32_f16<<<(65536 + 255) / 256, 256, 0, stream>>>(wq, wq16, 65536);
    cvt_f32_f16<<<(65536 + 255) / 256, 256, 0, stream>>>(wk, wk16, 65536);
    cvt_f32_f16<<<(65536 + 255) / 256, 256, 0, stream>>>(wv, wv16, 65536);
    cvt_f32_f16<<<(262144 + 255) / 256, 256, 0, stream>>>(w1, w116, 262144);
    cvt_f32_f16<<<(262144 + 255) / 256, 256, 0, stream>>>(w2, w216, 262144);

    // 2) QKV projections (WMMA): y[m][co] = x[m][:] . w[co][:] + b[co]
    dim3 gqkv(MROWS / 256, CDIM / 64);
    gemm_wmma<<<gqkv, 256, 0, stream>>>(x16, wq16, bq, yq, nullptr, CDIM, CDIM, 0);
    gemm_wmma<<<gqkv, 256, 0, stream>>>(x16, wk16, bk, yk, nullptr, CDIM, CDIM, 0);
    gemm_wmma<<<gqkv, 256, 0, stream>>>(x16, wv16, bv, yv, nullptr, CDIM, CDIM, 0);

    // 3) rotary embed q/k + scatter to [h][n][s][d]; v transposed to [h][n][d][s]
    rotary_scatter<<<(HEADS * NTOK * S_LEN * DHEAD) / 256, 256, 0, stream>>>(
        yq, yk, yv, qbuf, kbuf, vTb);

    // 4) attention (scores + decay mask -> softmax -> PV), 128KB LDS per block
    dim3 ga(S_LEN / 64, NTOK, HEADS);
    attn_wmma<<<ga, 256, 64 * S_LEN * sizeof(float), stream>>>(qbuf, kbuf, vTb, attn);

    // 5) LN1: (attn + x) -> ln1 (f32 for residual, f16 for FFN GEMM)
    ln_kernel<<<MROWS / 8, 256, 0, stream>>>(attn, x, ln1w, ln1b, ln1f, ln1h, 0);

    // 6) FFN GEMM1 with ReLU: h1 = relu(ln1 . w1^T + b1)
    dim3 gf1(MROWS / 256, FDIM / 64);
    gemm_wmma<<<gf1, 256, 0, stream>>>(ln1h, w116, b1, h1, nullptr, CDIM, FDIM, 1);

    // 7) FFN GEMM2: ffn = h1 . w2^T + b2
    dim3 gf2(MROWS / 256, CDIM / 64);
    gemm_wmma<<<gf2, 256, 0, stream>>>(h1, w216, b2, nullptr, ffn, FDIM, CDIM, 0);

    // 8) LN2: (ffn + ln1) -> layernorm -> transpose back to [B,N,C] f32 output
    ln_kernel<<<MROWS / 8, 256, 0, stream>>>(ffn, ln1f, ln2w, ln2b, (float*)d_out, nullptr, 1);
}
